// DynamicRadiusChannelFusion_70574902608063
// MI455X (gfx1250) — matrix operand order repaired and checked by reading, not compiled, not run on hardware
//
#include <hip/hip_runtime.h>
#include <hip/hip_bf16.h>
#include <math.h>

// Problem dimensions (fixed by the reference)
#define BB 4
#define NN 16384
#define MM 4096
#define KK 16
#define CC 128
#define OO 256

typedef _Float16 v16h __attribute__((ext_vector_type(16)));
typedef float    v8f  __attribute__((ext_vector_type(8)));

union Frag { uint4 u4[2]; v16h v; };

__device__ __forceinline__ unsigned long long shfl_xor_u64(unsigned long long v, int mask) {
    unsigned lo = (unsigned)v, hi = (unsigned)(v >> 32);
    lo = __shfl_xor(lo, mask, 32);
    hi = __shfl_xor(hi, mask, 32);
    return ((unsigned long long)hi << 32) | lo;
}

// ---------------------------------------------------------------------------
// Kernel 0: pack W1 (256x128), W2 (128x128), Wm (128x256) f32 -> f16 in
// WMMA B-fragment-linear layout: dst[tile*512 + lane*16 + e], where for
// tile (kt,nt): col=lane%16, h=lane/16; e<8 -> K=kt*32+8h+e ;
// e>=8 -> K=kt*32+16+8h+(e-8).  One contiguous 32B load per lane per fragment at use time.
// ---------------------------------------------------------------------------
__global__ __launch_bounds__(256) void pack_kernel(
    const float* __restrict__ W1, const float* __restrict__ W2,
    const float* __restrict__ Wm,
    _Float16* __restrict__ W1p, _Float16* __restrict__ W2p,
    _Float16* __restrict__ Wmp)
{
    int tid = blockIdx.x * 256 + threadIdx.x;
    const float* W; _Float16* D; int Ndim; int e;
    if (tid < 32768)      { W = W1; D = W1p; Ndim = 128; e = tid; }
    else if (tid < 49152) { W = W2; D = W2p; Ndim = 128; e = tid - 32768; }
    else if (tid < 81920) { W = Wm; D = Wmp; Ndim = 256; e = tid - 49152; }
    else return;
    int tile = e >> 9, within = e & 511;
    int lane = within >> 4, el = within & 15;
    int numNT = Ndim >> 4;
    int kt = tile / numNT, nt = tile % numNT;
    int hB = lane >> 4;
    int k = kt * 32 + 8 * hB + (el & 7) + ((el >> 3) << 4);
    int n = nt * 16 + (lane & 15);
    D[e] = (_Float16)W[k * Ndim + n];
}

// ---------------------------------------------------------------------------
// Kernel 1: radius-masked KNN (k=16).  One block of 256 threads per center.
// Each thread keeps a register-resident sorted top-16 over its 64-point
// stripe; merge via 16-round tournament on u64 keys (dist_bits<<32 | idx).
// ---------------------------------------------------------------------------
__global__ __launch_bounds__(256) void knn_kernel(
    const float* __restrict__ points, const int* __restrict__ center_idx,
    int* __restrict__ knn_out)
{
    int center = blockIdx.x;              // 0 .. B*M-1
    int b = center / MM;
    int tid = threadIdx.x;

    __shared__ float sd[256 * KK];
    __shared__ int   si[256 * KK];
    __shared__ unsigned long long warr[8];
    __shared__ unsigned long long bmin_s;

    int cidx = center_idx[center];
    const float* pb = points + (size_t)b * NN * 3;
    float cx = pb[cidx * 3 + 0], cy = pb[cidx * 3 + 1], cz = pb[cidx * 3 + 2];

    float dlist[KK]; int ilist[KK];
#pragma unroll
    for (int j = 0; j < KK; ++j) { dlist[j] = 3.0e38f; ilist[j] = 0x7fffffff; }

    for (int i = tid; i < NN; i += 256) {
        float dx = pb[i * 3 + 0] - cx;
        float dy = pb[i * 3 + 1] - cy;
        float dz = pb[i * 3 + 2] - cz;
        float dist = sqrtf(dx * dx + dy * dy + dz * dz + 1e-6f);
        float md = (dist <= 0.3f) ? dist : 1e9f;
        if (md < dlist[KK - 1]) {
            dlist[KK - 1] = md; ilist[KK - 1] = i;
#pragma unroll
            for (int j = KK - 1; j > 0; --j) {
                if (dlist[j] < dlist[j - 1]) {
                    float td = dlist[j]; dlist[j] = dlist[j - 1]; dlist[j - 1] = td;
                    int ti = ilist[j]; ilist[j] = ilist[j - 1]; ilist[j - 1] = ti;
                }
            }
        }
    }
#pragma unroll
    for (int j = 0; j < KK; ++j) { sd[tid * KK + j] = dlist[j]; si[tid * KK + j] = ilist[j]; }
    __syncthreads();

    int p = 0;
    int lane = tid & 31, wv = tid >> 5;
    for (int r = 0; r < KK; ++r) {
        unsigned long long key;
        if (p < KK) {
            unsigned ub = __float_as_uint(sd[tid * KK + p]);
            key = ((unsigned long long)ub << 32) | (unsigned)si[tid * KK + p];
        } else {
            key = ~0ull;
        }
        unsigned long long k2 = key;
#pragma unroll
        for (int m = 16; m > 0; m >>= 1) {
            unsigned long long o = shfl_xor_u64(k2, m);
            if (o < k2) k2 = o;
        }
        if (lane == 0) warr[wv] = k2;
        __syncthreads();
        if (tid == 0) {
            unsigned long long bm = warr[0];
#pragma unroll
            for (int j = 1; j < 8; ++j) if (warr[j] < bm) bm = warr[j];
            bmin_s = bm;
        }
        __syncthreads();
        unsigned long long bm = bmin_s;
        if (key == bm) ++p;
        if (tid == 0) knn_out[center * KK + r] = (int)(unsigned)(bm & 0xffffffffull);
        __syncthreads();
    }
}

// ---------------------------------------------------------------------------
// Kernel 2: fused gather + LN1 + GEMM1(256->128)+ReLU + GEMM2(128->128)
// + sigmoid + weighted K-mean + residual.  One block (8 wave32) per center;
// wave w owns output columns [16w, 16w+16).  All matrix math on WMMA f16.
// ---------------------------------------------------------------------------
__global__ __launch_bounds__(256) void mlp_kernel(
    const float* __restrict__ feats, const int* __restrict__ center_idx,
    const int* __restrict__ knn,
    const float* __restrict__ ln1w, const float* __restrict__ ln1b,
    const _Float16* __restrict__ W1p, const float* __restrict__ b1,
    const _Float16* __restrict__ W2p, const float* __restrict__ b2,
    const float* __restrict__ res_scale_p,
    _Float16* __restrict__ fusedh)
{
    int center = blockIdx.x;
    int b = center / MM;
    int tid = threadIdx.x;

    __shared__ int   kidx[KK];
    __shared__ float cf[CC];
    __shared__ float neigh[KK * CC];
    __shared__ __attribute__((aligned(16))) _Float16 Ab[KK * 2 * CC];  // 16x256
    __shared__ __attribute__((aligned(16))) _Float16 Hb[KK * CC];      // 16x128

    if (tid < KK) kidx[tid] = knn[center * KK + tid];
    int cidx = center_idx[center];
    const float* fb = feats + (size_t)b * NN * CC;
    if (tid < CC) cf[tid] = fb[(size_t)cidx * CC + tid];
    __syncthreads();
    for (int e = tid; e < KK * CC; e += 256) {
        int r = e >> 7, c = e & (CC - 1);
        neigh[e] = fb[(size_t)kidx[r] * CC + c];
    }
    __syncthreads();

    // LayerNorm over each 256-wide combo row (row = k-neighbor index).
    int row = tid >> 4, seg = tid & 15;
    float s = 0.f, ss = 0.f;
#pragma unroll
    for (int j = 0; j < 16; ++j) {
        int c = seg + 16 * j;
        float v = (c < CC) ? cf[c] : neigh[row * CC + (c - CC)];
        s += v; ss += v * v;
    }
#pragma unroll
    for (int m = 8; m > 0; m >>= 1) { s += __shfl_xor(s, m, 16); ss += __shfl_xor(ss, m, 16); }
    float mu = s * (1.f / 256.f);
    float var = ss * (1.f / 256.f) - mu * mu;
    float rstd = rsqrtf(var + 1e-5f);
#pragma unroll
    for (int j = 0; j < 16; ++j) {
        int c = seg + 16 * j;
        float v = (c < CC) ? cf[c] : neigh[row * CC + (c - CC)];
        Ab[row * (2 * CC) + c] = (_Float16)((v - mu) * rstd * ln1w[c] + ln1b[c]);
    }
    __syncthreads();

    int l = tid & 31, wv = tid >> 5;
    int mrow = l & 15, h = l >> 4;

    // GEMM1: (16x256) @ (256x128), wave wv -> cols [16wv, 16wv+16)
    v8f acc = {};
#pragma unroll
    for (int kt = 0; kt < 8; ++kt) {
        int k0 = kt * 32;
        Frag af, bf;
        af.u4[0] = *reinterpret_cast<const uint4*>(&Ab[mrow * 256 + k0 + 8 * h]);
        af.u4[1] = *reinterpret_cast<const uint4*>(&Ab[mrow * 256 + k0 + 16 + 8 * h]);
        const uint4* bp = reinterpret_cast<const uint4*>(W1p + (kt * 8 + wv) * 512 + l * 16);
        bf.u4[0] = bp[0]; bf.u4[1] = bp[1];
        acc = __builtin_amdgcn_wmma_f32_16x16x32_f16(false, af.v, false, bf.v,
                                                     (short)0, acc, false, false);
    }
    int col = wv * 16 + mrow;
    float bb1 = b1[col];
#pragma unroll
    for (int r = 0; r < 8; ++r) {
        float hv = acc[r] + bb1;
        Hb[(r + 8 * h) * CC + col] = (_Float16)(hv > 0.f ? hv : 0.f);
    }
    __syncthreads();

    // GEMM2: (16x128) @ (128x128)
    v8f acc2 = {};
#pragma unroll
    for (int kt = 0; kt < 4; ++kt) {
        int k0 = kt * 32;
        Frag af, bf;
        af.u4[0] = *reinterpret_cast<const uint4*>(&Hb[mrow * CC + k0 + 8 * h]);
        af.u4[1] = *reinterpret_cast<const uint4*>(&Hb[mrow * CC + k0 + 16 + 8 * h]);
        const uint4* bp = reinterpret_cast<const uint4*>(W2p + (kt * 8 + wv) * 512 + l * 16);
        bf.u4[0] = bp[0]; bf.u4[1] = bp[1];
        acc2 = __builtin_amdgcn_wmma_f32_16x16x32_f16(false, af.v, false, bf.v,
                                                      (short)0, acc2, false, false);
    }
    float bb2 = b2[col];
    float partial = 0.f;
#pragma unroll
    for (int r = 0; r < 8; ++r) {
        float x = acc2[r] + bb2;
        float w = 1.f / (1.f + __expf(-x));
        partial += w * neigh[(r + 8 * h) * CC + col];
    }
    partial += __shfl_xor(partial, 16, 32);   // sum rows 0..7 + 8..15
    if (h == 0) {
        float mean = partial * (1.f / (float)KK);
        float fused = cf[col] + res_scale_p[0] * mean;
        fusedh[(size_t)center * CC + col] = (_Float16)fused;
    }
}

// ---------------------------------------------------------------------------
// Kernel 3: out = LN2(relu(fused @ Wm + bm)).  16 centers per block;
// wave wv does 2 column tiles (cols 32wv..32wv+31).
// ---------------------------------------------------------------------------
__global__ __launch_bounds__(256) void out_kernel(
    const _Float16* __restrict__ fusedh, const _Float16* __restrict__ Wmp,
    const float* __restrict__ bm,
    const float* __restrict__ ln2w, const float* __restrict__ ln2b,
    float* __restrict__ out)
{
    int c0 = blockIdx.x * 16;
    int tid = threadIdx.x;
    __shared__ __attribute__((aligned(16))) _Float16 Ab[16 * CC];
    __shared__ float ob[16 * OO];

    for (int e = tid; e < 16 * CC; e += 256) Ab[e] = fusedh[(size_t)c0 * CC + e];
    __syncthreads();

    int l = tid & 31, wv = tid >> 5;
    int mrow = l & 15, h = l >> 4;
#pragma unroll
    for (int t = 0; t < 2; ++t) {
        int nt = wv * 2 + t;
        v8f acc = {};
#pragma unroll
        for (int kt = 0; kt < 4; ++kt) {
            int k0 = kt * 32;
            Frag af, bf;
            af.u4[0] = *reinterpret_cast<const uint4*>(&Ab[mrow * CC + k0 + 8 * h]);
            af.u4[1] = *reinterpret_cast<const uint4*>(&Ab[mrow * CC + k0 + 16 + 8 * h]);
            const uint4* bp = reinterpret_cast<const uint4*>(Wmp + (kt * 16 + nt) * 512 + l * 16);
            bf.u4[0] = bp[0]; bf.u4[1] = bp[1];
            acc = __builtin_amdgcn_wmma_f32_16x16x32_f16(false, af.v, false, bf.v,
                                                         (short)0, acc, false, false);
        }
        int col = nt * 16 + mrow;
        float bmc = bm[col];
#pragma unroll
        for (int r = 0; r < 8; ++r) {
            float v = acc[r] + bmc;
            ob[(r + 8 * h) * OO + col] = v > 0.f ? v : 0.f;
        }
    }
    __syncthreads();

    int row = tid >> 4, seg = tid & 15;
    float s = 0.f, ss = 0.f;
#pragma unroll
    for (int j = 0; j < 16; ++j) {
        float v = ob[row * OO + seg + 16 * j];
        s += v; ss += v * v;
    }
#pragma unroll
    for (int m = 8; m > 0; m >>= 1) { s += __shfl_xor(s, m, 16); ss += __shfl_xor(ss, m, 16); }
    float mu = s * (1.f / 256.f);
    float var = ss * (1.f / 256.f) - mu * mu;
    float rstd = rsqrtf(var + 1e-5f);
#pragma unroll
    for (int j = 0; j < 16; ++j) {
        int c = seg + 16 * j;
        float v = ob[row * OO + c];
        out[((size_t)(c0 + row)) * OO + c] = (v - mu) * rstd * ln2w[c] + ln2b[c];
    }
}

// ---------------------------------------------------------------------------
extern "C" void kernel_launch(void* const* d_in, const int* in_sizes, int n_in,
                              void* d_out, int out_size, void* d_ws, size_t ws_size,
                              hipStream_t stream) {
    (void)in_sizes; (void)n_in; (void)out_size; (void)ws_size;

    const float* points     = (const float*)d_in[0];
    const float* feats      = (const float*)d_in[1];
    const int*   center_idx = (const int*)d_in[2];
    const float* ln1_w      = (const float*)d_in[3];
    const float* ln1_b      = (const float*)d_in[4];
    const float* W1         = (const float*)d_in[5];
    const float* b1         = (const float*)d_in[6];
    const float* W2         = (const float*)d_in[7];
    const float* b2         = (const float*)d_in[8];
    const float* Wm         = (const float*)d_in[9];
    const float* bm         = (const float*)d_in[10];
    const float* ln2_w      = (const float*)d_in[11];
    const float* ln2_b      = (const float*)d_in[12];
    const float* res_scale  = (const float*)d_in[13];

    float* out = (float*)d_out;
    int*   knn = (int*)(out + (size_t)BB * MM * OO);   // knn_idx tail of d_out

    char* ws = (char*)d_ws;
    _Float16* W1p    = (_Float16*)(ws);                 //  64 KB
    _Float16* W2p    = (_Float16*)(ws + 65536);         //  32 KB
    _Float16* Wmp    = (_Float16*)(ws + 98304);         //  64 KB
    _Float16* fusedh = (_Float16*)(ws + 163840);        //   4 MB

    pack_kernel<<<(81920 + 255) / 256, 256, 0, stream>>>(W1, W2, Wm, W1p, W2p, Wmp);
    knn_kernel<<<BB * MM, 256, 0, stream>>>(points, center_idx, knn);
    mlp_kernel<<<BB * MM, 256, 0, stream>>>(feats, center_idx, knn,
                                            ln1_w, ln1_b, W1p, b1, W2p, b2,
                                            res_scale, fusedh);
    out_kernel<<<(BB * MM) / 16, 256, 0, stream>>>(fusedh, Wmp, bm, ln2_w, ln2_b, out);
}
